// DecLayer_33852932227487
// MI455X (gfx1250) — matrix operand order repaired and checked by reading, not compiled, not run on hardware
//
#include <hip/hip_runtime.h>
#include <math.h>

// ---------------------------------------------------------------------------
// DecLayer for MI455X (gfx1250): fused message-MLP + aggregate + LN + FFN + LN
// B=4, L=2048, K=48, H=128, NIN=512. f16 WMMA (16x16x32), f32 accumulate.
// - Activations staged in LDS (A fragments = 2x ds_load_b128, row-major).
// - Weights pre-packed to WMMA-fragment order in global memory and read
//   directly from the 192MB L2 (B fragment = 2x global_load_b128): each B
//   element is consumed once per workgroup, so LDS staging would only double
//   the byte movement and cost 196KB of LDS (occupancy 1 WG/WGP -> 4 WG/WGP).
// - K-outer / tile-inner loops reuse one operand across 3-4 WMMAs in registers.
// ---------------------------------------------------------------------------

#define HDIM   128
#define KNBR   48
#define NINDIM 512
#define NEDIM  384          // NIN - H
#define SCALE_INV (1.0f / 30.0f)
#define LN_EPS 1e-5f

typedef _Float16 v16h __attribute__((ext_vector_type(16)));
typedef _Float16 v8h  __attribute__((ext_vector_type(8)));
typedef _Float16 v4h  __attribute__((ext_vector_type(4)));
typedef float    v8f  __attribute__((ext_vector_type(8)));

// Branch-free GELU via hardware transcendentals:
// 0.5*x*(1+tanh(u)) == x*sigmoid(2u), u = sqrt(2/pi)*(x + 0.044715 x^3).
__device__ __forceinline__ float gelu_f(float x) {
    const float u = x * (0.7978845608028654f + 0.03567740814183479f * x * x);
    const float e = __builtin_amdgcn_exp2f(-2.8853900817779268f * u); // exp(-2u)
    return x * __builtin_amdgcn_rcpf(1.0f + e);
}

__device__ __forceinline__ float wave_sum(float x) {
    #pragma unroll
    for (int off = 16; off > 0; off >>= 1) x += __shfl_xor(x, off, 32);
    return x;
}

// A-matrix fragment, 16x32 f16 (MxK), from row-major LDS [rows x lda].
// Lane's 16 halfs are k = 16*(lane>>4) + 0..15, contiguous p[0..15]
// (16B-aligned) -> 2x ds_load_b128.
__device__ __forceinline__ v16h load_a16x32(const _Float16* S, int lda,
                                            int r0, int k0, int lane) {
    const _Float16* p = S + (r0 + (lane & 15)) * lda + k0 + ((lane >> 4) << 4);
    v8h lo = *(const v8h*)(p);
    v8h hi = *(const v8h*)(p + 8);
    return __builtin_shufflevector(lo, hi, 0, 1, 2, 3, 4, 5, 6, 7,
                                   8, 9, 10, 11, 12, 13, 14, 15);
}

// B-matrix fragment, 32x16 f16 (KxN), from GLOBAL memory pre-packed in
// fragment order: tile = kt*(NN/16)+ct, lane-major 32 lanes x 16 halfs.
// L2-resident -> 2x global_load_b128 per fragment.
__device__ __forceinline__ v16h load_b_frag(const _Float16* __restrict__ P,
                                            int tile, int lane) {
    const _Float16* p = P + tile * 512 + lane * 16;   // 32B per lane, aligned
    v8h lo = *(const v8h*)(p);
    v8h hi = *(const v8h*)(p + 8);
    return __builtin_shufflevector(lo, hi, 0, 1, 2, 3, 4, 5, 6, 7,
                                   8, 9, 10, 11, 12, 13, 14, 15);
}

__device__ __forceinline__ v8f wmma_f16(v16h a, v16h b, v8f c) {
    return __builtin_amdgcn_wmma_f32_16x16x32_f16(false, a, false, b,
                                                  (short)0, c, false, false);
}

// ---------------------------------------------------------------------------
// Pack kernel: f32 row-major [K x NN] weight -> f16 WMMA-fragment order.
// Element (k,n): tile=(k/32)*(NN/16)+(n/16); lane=(n&15)|(((k>>4)&1)<<4); h=k&15.
// ---------------------------------------------------------------------------
__global__ void __launch_bounds__(256) pack_kernel(
    const float* __restrict__ src, unsigned short* __restrict__ dst,
    int NN, int total)
{
    const int i = blockIdx.x * 256 + threadIdx.x;
    if (i >= total) return;
    const int k = i / NN, n = i % NN;
    const int tile = (k >> 5) * (NN >> 4) + (n >> 4);
    const int lane = (n & 15) | (((k >> 4) & 1) << 4);
    union { _Float16 h; unsigned short u; } cv;
    cv.h = (_Float16)src[i];
    dst[tile * 512 + lane * 16 + (k & 15)] = cv.u;
}

// ---------------------------------------------------------------------------
// Kernel 1: per-node message MLP + masked aggregation.  grid = B*L, block = 256.
// LDS = 74KB -> 4 workgroups per WGP.
// ---------------------------------------------------------------------------
__global__ void __launch_bounds__(256) msg_kernel(
    const float* __restrict__ hV, const float* __restrict__ hE,
    const float* __restrict__ maskE,
    const _Float16* __restrict__ pw1, const float* __restrict__ b1,
    const _Float16* __restrict__ pw2, const float* __restrict__ b2,
    const _Float16* __restrict__ pw3, const float* __restrict__ b3,
    float* __restrict__ dh)
{
    extern __shared__ char smem[];
    _Float16* sEV = (_Float16*)smem;          // 48 x 512 row-major
    _Float16* sH1 = sEV + KNBR * NINDIM;      // 48 x 128 row-major
    _Float16* sH2 = sH1 + KNBR * HDIM;        // 48 x 128 row-major
    float*    sM  = (float*)(sH2 + KNBR * HDIM); // 48

    const int node = blockIdx.x;
    const int tid  = threadIdx.x;
    const int lane = tid & 31;
    const int wave = tid >> 5;

    // ---- stage concatenated [hV | hE] rows: float4 loads, v4h stores ----
    const float* hVrow = hV + (size_t)node * HDIM;
    const float* hErow = hE + (size_t)node * KNBR * NEDIM;
    for (int i4 = tid; i4 < (KNBR * NINDIM) / 4; i4 += 256) {
        const int i = i4 * 4;
        const int k = i >> 9;             // /512
        const int c = i & (NINDIM - 1);
        const float4 f = (c < HDIM) ? *(const float4*)(hVrow + c)
                                    : *(const float4*)(hErow + k * NEDIM + (c - HDIM));
        v4h h = { (_Float16)f.x, (_Float16)f.y, (_Float16)f.z, (_Float16)f.w };
        *(v4h*)(sEV + i) = h;
    }
    if (tid < KNBR) sM[tid] = maskE[(size_t)node * KNBR + tid];
    __syncthreads();

    const int n0    = wave * 16;          // this wave's output column strip
    const int nlane = lane & 15;
    const int mhalf = (lane >> 4) << 3;   // 0 or 8

    // ---- GEMM1: [48x512] @ [512x128]; B from L2, shared across 3 row tiles ----
    {
        v8f acc0 = {}, acc1 = {}, acc2 = {};
        #pragma unroll 2
        for (int kk = 0; kk < 16; ++kk) {
            const v16h b = load_b_frag(pw1, kk * 8 + wave, lane);
            acc0 = wmma_f16(load_a16x32(sEV, NINDIM,  0, kk * 32, lane), b, acc0);
            acc1 = wmma_f16(load_a16x32(sEV, NINDIM, 16, kk * 32, lane), b, acc1);
            acc2 = wmma_f16(load_a16x32(sEV, NINDIM, 32, kk * 32, lane), b, acc2);
        }
        const float bias = b1[n0 + nlane];
        #pragma unroll
        for (int v = 0; v < 8; ++v) {
            const int m = mhalf + v;
            sH1[(m)      * HDIM + n0 + nlane] = (_Float16)gelu_f(acc0[v] + bias);
            sH1[(m + 16) * HDIM + n0 + nlane] = (_Float16)gelu_f(acc1[v] + bias);
            sH1[(m + 32) * HDIM + n0 + nlane] = (_Float16)gelu_f(acc2[v] + bias);
        }
    }
    __syncthreads();

    // ---- GEMM2: [48x128] @ [128x128], bias + gelu -> sH2 ----
    {
        v8f acc0 = {}, acc1 = {}, acc2 = {};
        #pragma unroll
        for (int kk = 0; kk < 4; ++kk) {
            const v16h b = load_b_frag(pw2, kk * 8 + wave, lane);
            acc0 = wmma_f16(load_a16x32(sH1, HDIM,  0, kk * 32, lane), b, acc0);
            acc1 = wmma_f16(load_a16x32(sH1, HDIM, 16, kk * 32, lane), b, acc1);
            acc2 = wmma_f16(load_a16x32(sH1, HDIM, 32, kk * 32, lane), b, acc2);
        }
        const float bias = b2[n0 + nlane];
        #pragma unroll
        for (int v = 0; v < 8; ++v) {
            const int m = mhalf + v;
            sH2[(m)      * HDIM + n0 + nlane] = (_Float16)gelu_f(acc0[v] + bias);
            sH2[(m + 16) * HDIM + n0 + nlane] = (_Float16)gelu_f(acc1[v] + bias);
            sH2[(m + 32) * HDIM + n0 + nlane] = (_Float16)gelu_f(acc2[v] + bias);
        }
    }
    __syncthreads();

    // ---- GEMM3: [48x128] @ [128x128] + b3, mask, reduce over the 48 rows ----
    {
        v8f acc0 = {}, acc1 = {}, acc2 = {};
        #pragma unroll
        for (int kk = 0; kk < 4; ++kk) {
            const v16h b = load_b_frag(pw3, kk * 8 + wave, lane);
            acc0 = wmma_f16(load_a16x32(sH2, HDIM,  0, kk * 32, lane), b, acc0);
            acc1 = wmma_f16(load_a16x32(sH2, HDIM, 16, kk * 32, lane), b, acc1);
            acc2 = wmma_f16(load_a16x32(sH2, HDIM, 32, kk * 32, lane), b, acc2);
        }
        const float bias = b3[n0 + nlane];
        float part = 0.0f;
        #pragma unroll
        for (int v = 0; v < 8; ++v) {
            const int m = mhalf + v;
            part += (acc0[v] + bias) * sM[m];
            part += (acc1[v] + bias) * sM[m + 16];
            part += (acc2[v] + bias) * sM[m + 32];
        }
        part += __shfl_xor(part, 16, 32);   // fold M upper-half lanes into lower
        if (lane < 16)
            dh[(size_t)node * HDIM + n0 + nlane] = part * SCALE_INV;
    }
}

// ---------------------------------------------------------------------------
// Kernel 2: residual + LN1 + FFN(128->512->128) + residual + LN2 + mask_V.
// grid = (B*L)/16, block = 256. 16 node-rows per workgroup. LDS = 37KB.
// ---------------------------------------------------------------------------
__global__ void __launch_bounds__(256) node_kernel(
    const float* __restrict__ hV, const float* __restrict__ dh,
    const float* __restrict__ maskV,
    const float* __restrict__ g1, const float* __restrict__ bn1,
    const float* __restrict__ g2, const float* __restrict__ bn2,
    const _Float16* __restrict__ pwf1, const float* __restrict__ bf1,
    const _Float16* __restrict__ pwf2, const float* __restrict__ bf2,
    float* __restrict__ out)
{
    extern __shared__ char smem[];
    _Float16* sXh = (_Float16*)smem;         // 16 x 128 (f16 LN1 output)
    _Float16* sF  = sXh + 16 * HDIM;         // 16 x 512 (f16 gelu(FFN1))
    float*    sXf = (float*)(sF + 16 * 512); // 16 x 128 (f32 residual)
    float*    sY  = sXf + 16 * HDIM;         // 16 x 128 (f32 pre-LN2)

    const int tid  = threadIdx.x;
    const int lane = tid & 31;
    const int wave = tid >> 5;
    const int row0 = blockIdx.x * 16;
    const int nlane = lane & 15;
    const int mhalf = (lane >> 4) << 3;

    // ---- x = h_V + dh ----
    for (int i = tid; i < 16 * HDIM; i += 256) {
        const int r = i >> 7, c = i & (HDIM - 1);
        const size_t g = (size_t)(row0 + r) * HDIM + c;
        sXf[i] = hV[g] + dh[g];
    }
    __syncthreads();

    // ---- LN1 (wave32 shuffle reductions; 2 rows per wave) ----
    #pragma unroll
    for (int rr = 0; rr < 2; ++rr) {
        const int r = wave + rr * 8;
        float x[4];
        #pragma unroll
        for (int i = 0; i < 4; ++i) x[i] = sXf[r * HDIM + lane * 4 + i];
        const float mean = wave_sum(x[0] + x[1] + x[2] + x[3]) * (1.0f / 128.0f);
        float vs = 0.0f;
        #pragma unroll
        for (int i = 0; i < 4; ++i) { const float d = x[i] - mean; vs += d * d; }
        const float inv = rsqrtf(wave_sum(vs) * (1.0f / 128.0f) + LN_EPS);
        #pragma unroll
        for (int i = 0; i < 4; ++i) {
            const int c = lane * 4 + i;
            const float y = (x[i] - mean) * inv * g1[c] + bn1[c];
            sXf[r * HDIM + c] = y;               // f32 residual
            sXh[r * HDIM + c] = (_Float16)y;     // f16 WMMA operand
        }
    }
    __syncthreads();

    // ---- FFN1: [16x128] @ [128x512]; A fragment shared across 4 col tiles ----
    {
        v8f a0 = {}, a1 = {}, a2 = {}, a3 = {};
        #pragma unroll
        for (int kk = 0; kk < 4; ++kk) {
            const v16h a = load_a16x32(sXh, HDIM, 0, kk * 32, lane);
            a0 = wmma_f16(a, load_b_frag(pwf1, kk * 32 + wave,      lane), a0);
            a1 = wmma_f16(a, load_b_frag(pwf1, kk * 32 + wave + 8,  lane), a1);
            a2 = wmma_f16(a, load_b_frag(pwf1, kk * 32 + wave + 16, lane), a2);
            a3 = wmma_f16(a, load_b_frag(pwf1, kk * 32 + wave + 24, lane), a3);
        }
        #pragma unroll
        for (int j = 0; j < 4; ++j) {
            const v8f acc = (j == 0) ? a0 : (j == 1) ? a1 : (j == 2) ? a2 : a3;
            const int n0 = (wave + 8 * j) * 16;
            const float bias = bf1[n0 + nlane];
            #pragma unroll
            for (int v = 0; v < 8; ++v)
                sF[(mhalf + v) * 512 + n0 + nlane] = (_Float16)gelu_f(acc[v] + bias);
        }
    }
    __syncthreads();

    // ---- FFN2: [16x512] @ [512x128] + bias + residual -> sY ----
    {
        const int n0 = wave * 16;
        v8f acc = {};
        #pragma unroll 4
        for (int kk = 0; kk < 16; ++kk)
            acc = wmma_f16(load_a16x32(sF, 512, 0, kk * 32, lane),
                           load_b_frag(pwf2, kk * 8 + wave, lane), acc);
        const float bias = bf2[n0 + nlane];
        #pragma unroll
        for (int v = 0; v < 8; ++v) {
            const int m = mhalf + v;
            sY[m * HDIM + n0 + nlane] = acc[v] + bias + sXf[m * HDIM + n0 + nlane];
        }
    }
    __syncthreads();

    // ---- LN2 + mask_V + store ----
    #pragma unroll
    for (int rr = 0; rr < 2; ++rr) {
        const int r = wave + rr * 8;
        float x[4];
        #pragma unroll
        for (int i = 0; i < 4; ++i) x[i] = sY[r * HDIM + lane * 4 + i];
        const float mean = wave_sum(x[0] + x[1] + x[2] + x[3]) * (1.0f / 128.0f);
        float vs = 0.0f;
        #pragma unroll
        for (int i = 0; i < 4; ++i) { const float d = x[i] - mean; vs += d * d; }
        const float inv = rsqrtf(wave_sum(vs) * (1.0f / 128.0f) + LN_EPS);
        const float mv = maskV[row0 + r];
        #pragma unroll
        for (int i = 0; i < 4; ++i) {
            const int c = lane * 4 + i;
            out[(size_t)(row0 + r) * HDIM + c] =
                mv * ((x[i] - mean) * inv * g2[c] + bn2[c]);
        }
    }
}

// ---------------------------------------------------------------------------
extern "C" void kernel_launch(void* const* d_in, const int* in_sizes, int n_in,
                              void* d_out, int out_size, void* d_ws, size_t ws_size,
                              hipStream_t stream) {
    (void)n_in; (void)out_size; (void)ws_size;
    const float* hV    = (const float*)d_in[0];
    const float* hE    = (const float*)d_in[1];
    const float* maskV = (const float*)d_in[2];
    const float* maskE = (const float*)d_in[3];
    const float* w1    = (const float*)d_in[4];
    const float* b1    = (const float*)d_in[5];
    const float* w2    = (const float*)d_in[6];
    const float* b2    = (const float*)d_in[7];
    const float* w3    = (const float*)d_in[8];
    const float* b3    = (const float*)d_in[9];
    const float* g1    = (const float*)d_in[10];
    const float* bn1   = (const float*)d_in[11];
    const float* g2    = (const float*)d_in[12];
    const float* bn2   = (const float*)d_in[13];
    const float* wf1   = (const float*)d_in[14];
    const float* bf1   = (const float*)d_in[15];
    const float* wf2   = (const float*)d_in[16];
    const float* bf2   = (const float*)d_in[17];
    float* out = (float*)d_out;

    const int BL = in_sizes[2];          // mask_V element count = B*L

    // workspace layout: [dh: BL*H f32][packed f16 weights, L2-resident]
    float* dh = (float*)d_ws;
    unsigned short* pw  = (unsigned short*)((char*)d_ws +
                          (size_t)BL * HDIM * sizeof(float));
    unsigned short* pw1  = pw;            // 512x128 -> 65536
    unsigned short* pw2  = pw1 + 65536;   // 128x128 -> 16384
    unsigned short* pw3  = pw2 + 16384;   // 128x128 -> 16384
    unsigned short* pwf1 = pw3 + 16384;   // 128x512 -> 65536
    unsigned short* pwf2 = pwf1 + 65536;  // 512x128 -> 65536

    // one-shot weight repack (f32 row-major -> f16 WMMA-fragment order)
    pack_kernel<<<(65536 + 255) / 256, 256, 0, stream>>>(w1,  pw1,  128, 65536);
    pack_kernel<<<(16384 + 255) / 256, 256, 0, stream>>>(w2,  pw2,  128, 16384);
    pack_kernel<<<(16384 + 255) / 256, 256, 0, stream>>>(w3,  pw3,  128, 16384);
    pack_kernel<<<(65536 + 255) / 256, 256, 0, stream>>>(wf1, pwf1, 512, 65536);
    pack_kernel<<<(65536 + 255) / 256, 256, 0, stream>>>(wf2, pwf2, 128, 65536);

    // kernel1 LDS = 73920 bytes -> 4 workgroups per WGP
    const size_t lds1 = (size_t)(KNBR * NINDIM + 2 * KNBR * HDIM) * 2 +
                        KNBR * sizeof(float);
    msg_kernel<<<BL, 256, lds1, stream>>>(
        hV, hE, maskE, (const _Float16*)pw1, b1, (const _Float16*)pw2, b2,
        (const _Float16*)pw3, b3, dh);

    // kernel2 LDS = 36864 bytes
    const size_t lds2 = (size_t)(16 * HDIM + 16 * 512) * 2 +
                        (size_t)(2 * 16 * HDIM) * sizeof(float);
    node_kernel<<<BL / 16, 256, lds2, stream>>>(
        hV, dh, maskV, g1, bn1, g2, bn2,
        (const _Float16*)pwf1, bf1, (const _Float16*)pwf2, bf2, out);
}
